// SkipDecoder_78176994722609
// MI455X (gfx1250) — matrix operand order repaired and checked by reading, not compiled, not run on hardware
//
#include <hip/hip_runtime.h>
#include <hip/hip_bf16.h>

// ---------------------------------------------------------------------------
// SkipDecoder for MI455X (gfx1250, wave32, WMMA).
//
// Reference facts exploited:
//  * h0 == 0 always  -> W_hh term vanishes (d_in[3] unused).
//  * per-step logits == final einsum output -> write logits directly to d_out.
//  * ctx part of x is step-invariant -> precompute "base" gates once.
// Heavy math runs through V_WMMA_F32_16X16X32_F16 (f16 in, f32 accumulate).
// Each wave computes 5 adjacent 16x16 col tiles sharing one A fragment:
// 12 b128 loads per 5 WMMAs (2.4 VMEM/WMMA instead of 4).
// ---------------------------------------------------------------------------

typedef __attribute__((ext_vector_type(16))) _Float16 v16h;
typedef __attribute__((ext_vector_type(8)))  float    v8f;
typedef __attribute__((ext_vector_type(4)))  float    f32x4;

#define B_SEQ   128        // decoder batch (B-2)
#define THOUGHT 1200
#define WORD    620
#define WORD_P  640        // WORD padded to multiple of 32 (WMMA K step)
#define VOCAB   20000
#define GATES   2480       // 4*WORD
#define CTXK    2400       // 2*THOUGHT (multiple of 32 already)
#define IN_COLS 3020       // 2*THOUGHT + WORD (W_ih leading dim)
#define MAXLEN  30
#define TN      5          // col tiles per wave; divides 2480/16=155 and 20000/16=1250

union F16Frag { v16h v; f32x4 q[2]; };

// out[m][n] = sum_k A[m][k]*W[n][k]  (+bias1[n]) (+bias2[n]) (+addmat[m][n])
// A: [128 x lda] f16 row-major, W: [N x ldw] f16 row-major, K multiple of 32.
// grid.x = N/(16*TN); block = 256 = 8 waves = 8 row tiles (M=128).
// All 8 waves share the block's 5 B col tiles -> B re-reads hit WGP$.
__global__ __launch_bounds__(256) void gemm16_nt(
    const _Float16* __restrict__ A, int lda,
    const _Float16* __restrict__ W, int ldw,
    float* __restrict__ out, long ldOut, long outOff, int K,
    const float* __restrict__ bias1, const float* __restrict__ bias2,
    const float* __restrict__ addmat, int ldAdd)
{
    const int wave = threadIdx.x >> 5;
    const int lane = threadIdx.x & 31;
    const int kh   = lane >> 4;        // K-half selector (ISA A/B layout)
    const int l16  = lane & 15;
    const int m0   = wave * 16;
    const long n0  = (long)blockIdx.x * (16 * TN);

    // A 16-bit 16x32 layout: lane m, half 0 -> K {0..7,16..23}; half 1 -> {8..15,24..31}
    const _Float16* aRow = A + (long)(m0 + l16) * lda + kh * 8;
    // B 16-bit 32x16 layout: lane n, half 0 -> K {0..7,16..23}; half 1 -> {8..15,24..31}
    const _Float16* bRow = W + (n0 + l16) * ldw + kh * 8;
    const long bTile = (long)16 * ldw;   // elements between adjacent col tiles

    v8f acc[TN];
    #pragma unroll
    for (int j = 0; j < TN; ++j) acc[j] = (v8f){};

    for (int kb = 0; kb < K; kb += 32) {
        F16Frag a;
        a.q[0] = *reinterpret_cast<const f32x4*>(aRow + kb);
        a.q[1] = *reinterpret_cast<const f32x4*>(aRow + kb + 16);
        #pragma unroll
        for (int j = 0; j < TN; ++j) {
            F16Frag b;
            const _Float16* bp = bRow + j * bTile + kb;
            b.q[0] = *reinterpret_cast<const f32x4*>(bp);
            b.q[1] = *reinterpret_cast<const f32x4*>(bp + 16);
            acc[j] = __builtin_amdgcn_wmma_f32_16x16x32_f16(
                /*neg_a=*/false, a.v, /*neg_b=*/false, b.v,
                /*c_mod=*/(short)0, acc[j], /*reuse_a=*/false, /*reuse_b=*/false);
        }
    }

    // C layout: VGPR r, lanes 0-15 -> M=r ; lanes 16-31 -> M=8+r ; N = l16
    #pragma unroll
    for (int j = 0; j < TN; ++j) {
        const int col = (int)n0 + j * 16 + l16;
        float extra = 0.f;
        if (bias1) extra += bias1[col];
        if (bias2) extra += bias2[col];
        #pragma unroll
        for (int r = 0; r < 8; ++r) {
            int row = m0 + r + 8 * kh;
            float v = acc[j][r] + extra;
            if (addmat) v += addmat[(long)row * ldAdd + col];
            out[(long)row * ldOut + outOff + col] = v;
        }
    }
}

// f32 -> f16 with zero padding of trailing columns.
__global__ void cvt_pad_f16(const float* __restrict__ src, int srcLd, int srcOff,
                            int srcCols, _Float16* __restrict__ dst, int dstLd,
                            long total)
{
    long i = (long)blockIdx.x * blockDim.x + threadIdx.x;
    if (i >= total) return;
    int r = (int)(i / dstLd);
    int c = (int)(i % dstLd);
    float v = (c < srcCols) ? src[(long)r * srcLd + srcOff + c] : 0.f;
    dst[i] = (_Float16)v;
}

// ctx16[n][j] = concat(thoughts[n], thoughts[n+2]) as f16
__global__ void build_ctx16(const float* __restrict__ thoughts,
                            _Float16* __restrict__ ctx16)
{
    long i = (long)blockIdx.x * blockDim.x + threadIdx.x;
    if (i >= (long)B_SEQ * CTXK) return;
    int n = (int)(i / CTXK);
    int j = (int)(i % CTXK);
    float v = (j < THOUGHT) ? thoughts[(long)n * THOUGHT + j]
                            : thoughts[(long)(n + 2) * THOUGHT + (j - THOUGHT)];
    ctx16[i] = (_Float16)v;
}

__global__ void zero_f16(_Float16* __restrict__ p, long total)
{
    long i = (long)blockIdx.x * blockDim.x + threadIdx.x;
    if (i < total) p[i] = (_Float16)0.f;
}

// h = sigmoid(o) * tanh( sigmoid(i) * tanh(g) )   (f-gate dead: c_prev == 0)
__global__ void lstm_act(const float* __restrict__ gates, _Float16* __restrict__ h16)
{
    int i = blockIdx.x * blockDim.x + threadIdx.x;
    if (i >= B_SEQ * WORD_P) return;
    int n = i / WORD_P, w = i % WORD_P;
    float v = 0.f;
    if (w < WORD) {
        const float* g = gates + (long)n * GATES;
        float ig = g[w];
        float gg = g[2 * WORD + w];
        float og = g[3 * WORD + w];
        float c  = (1.f / (1.f + __expf(-ig))) * tanhf(gg);
        v        = (1.f / (1.f + __expf(-og))) * tanhf(c);
    }
    h16[i] = (_Float16)v;
}

// Per-row argmax over the logits just written to d_out, then gather the
// greedy-decoded word embedding (as f16, zero-padded) into last16.
__global__ __launch_bounds__(256) void argmax_embed(
    const float* __restrict__ out, long ldOut, long off,
    const float* __restrict__ embed, _Float16* __restrict__ last16)
{
    __shared__ float sv[256];
    __shared__ int   si[256];
    const int n = blockIdx.x;
    const int tid = threadIdx.x;
    const float* row = out + (long)n * ldOut + off;

    float best = -3.402823466e38f;
    int   bi   = 0x7FFFFFFF;
    for (int v = tid; v < VOCAB; v += 256) {
        float x = row[v];
        if (x > best) { best = x; bi = v; }   // strict > keeps first max
    }
    sv[tid] = best; si[tid] = bi;
    __syncthreads();
    for (int s = 128; s > 0; s >>= 1) {
        if (tid < s) {
            float ov = sv[tid + s]; int oi = si[tid + s];
            if (ov > sv[tid] || (ov == sv[tid] && oi < si[tid])) {
                sv[tid] = ov; si[tid] = oi;
            }
        }
        __syncthreads();
    }
    const long idx = si[0];
    const float* e = embed + idx * WORD;
    for (int w = tid; w < WORD_P; w += 256)
        last16[(long)n * WORD_P + w] = (w < WORD) ? (_Float16)e[w] : (_Float16)0.f;
}

extern "C" void kernel_launch(void* const* d_in, const int* in_sizes, int n_in,
                              void* d_out, int out_size, void* d_ws, size_t ws_size,
                              hipStream_t stream)
{
    (void)in_sizes; (void)n_in; (void)out_size; (void)ws_size;
    const float* thoughts = (const float*)d_in[0];
    const float* embed    = (const float*)d_in[1];
    const float* W_ih     = (const float*)d_in[2];
    // d_in[3] = W_hh is dead: reference multiplies it by an all-zero h0.
    const float* b_ih     = (const float*)d_in[4];
    const float* b_hh     = (const float*)d_in[5];
    const float* Ww       = (const float*)d_in[6];
    const float* bw       = (const float*)d_in[7];
    float* out = (float*)d_out;

    // Carve workspace (all buffers fully rewritten each call; ~42 MiB total).
    char* p = (char*)d_ws;
    auto carve = [&](size_t bytes) -> char* {
        char* r = p; p += (bytes + 255) & ~(size_t)255; return r;
    };
    _Float16* Ww16   = (_Float16*)carve((size_t)VOCAB * WORD_P * 2);  // [20000][640]
    _Float16* Wic16  = (_Float16*)carve((size_t)GATES * CTXK   * 2);  // [2480][2400]
    _Float16* Wiw16  = (_Float16*)carve((size_t)GATES * WORD_P * 2);  // [2480][640]
    _Float16* ctx16  = (_Float16*)carve((size_t)B_SEQ * CTXK   * 2);  // [128][2400]
    float*    base   = (float*)   carve((size_t)B_SEQ * GATES  * 4);  // [128][2480]
    float*    gates  = (float*)   carve((size_t)B_SEQ * GATES  * 4);  // [128][2480]
    _Float16* h16    = (_Float16*)carve((size_t)B_SEQ * WORD_P * 2);  // [128][640]
    _Float16* last16 = (_Float16*)carve((size_t)B_SEQ * WORD_P * 2);  // [128][640]

    auto nblk = [](long total, int bs) { return (int)((total + bs - 1) / bs); };

    // One-time (per launch) conversions to f16 with K padding.
    cvt_pad_f16<<<nblk((long)VOCAB * WORD_P, 256), 256, 0, stream>>>(
        Ww, WORD, 0, WORD, Ww16, WORD_P, (long)VOCAB * WORD_P);
    cvt_pad_f16<<<nblk((long)GATES * CTXK, 256), 256, 0, stream>>>(
        W_ih, IN_COLS, 0, CTXK, Wic16, CTXK, (long)GATES * CTXK);
    cvt_pad_f16<<<nblk((long)GATES * WORD_P, 256), 256, 0, stream>>>(
        W_ih, IN_COLS, CTXK, WORD, Wiw16, WORD_P, (long)GATES * WORD_P);
    build_ctx16<<<nblk((long)B_SEQ * CTXK, 256), 256, 0, stream>>>(thoughts, ctx16);
    zero_f16<<<nblk((long)B_SEQ * WORD_P, 256), 256, 0, stream>>>(
        last16, (long)B_SEQ * WORD_P);  // step 0: last_word = 0

    // base = ctx @ W_ih[:, :2400].T + b_ih + b_hh   (step-invariant gates part)
    gemm16_nt<<<GATES / (16 * TN), 256, 0, stream>>>(
        ctx16, CTXK, Wic16, CTXK, base, GATES, 0, CTXK, b_ih, b_hh, nullptr, 0);

    for (int t = 0; t < MAXLEN; ++t) {
        // gates = base + last_word @ W_ih[:, 2400:].T
        gemm16_nt<<<GATES / (16 * TN), 256, 0, stream>>>(
            last16, WORD_P, Wiw16, WORD_P, gates, GATES, 0, WORD_P,
            nullptr, nullptr, base, GATES);
        lstm_act<<<nblk(B_SEQ * WORD_P, 256), 256, 0, stream>>>(gates, h16);
        // logits for step t == final output slice [*, t, :]  (written once)
        gemm16_nt<<<VOCAB / (16 * TN), 256, 0, stream>>>(
            h16, WORD_P, Ww16, WORD_P, out,
            (long)MAXLEN * VOCAB, (long)t * VOCAB, WORD_P,
            bw, nullptr, nullptr, 0);
        if (t + 1 < MAXLEN)  // last step's decode is never consumed
            argmax_embed<<<B_SEQ, 256, 0, stream>>>(
                out, (long)MAXLEN * VOCAB, (long)t * VOCAB, embed, last16);
    }
}
